// FeedFoward_52518860095615
// MI455X (gfx1250) — 8-bit matrix operand order repaired and checked by reading, compiled, not run on hardware
//
#include <hip/hip_runtime.h>
#include <cstdint>
#include <cmath>


typedef int v8i __attribute__((ext_vector_type(8)));
typedef int v4i __attribute__((ext_vector_type(4)));

#define EPSQ 1e-5f
#define NPART 512

#if defined(__has_builtin)
#  if __has_builtin(__builtin_amdgcn_global_load_async_to_lds_b128)
#    define USE_ASYNC_LDS 1
#  endif
#endif
#ifndef USE_ASYNC_LDS
#  define USE_ASYNC_LDS 0
#endif

#if USE_ASYNC_LDS
#define AS1V4(p) ((__attribute__((address_space(1))) v4i*)(p))
#define AS3V4(p) ((__attribute__((address_space(3))) v4i*)(p))
static __device__ __forceinline__ void wait_async0() {
#if __has_builtin(__builtin_amdgcn_s_wait_asynccnt)
  __builtin_amdgcn_s_wait_asynccnt(0);
#else
  asm volatile("s_wait_asynccnt 0x0" ::: "memory");
#endif
}
#endif

__global__ __launch_bounds__(256) void sumsq_partial(const float* __restrict__ w,
                                                     long long n,
                                                     float* __restrict__ partial) {
  __shared__ float red[256];
  float s = 0.f;
  for (long long i = (long long)blockIdx.x * 256 + threadIdx.x; i < n;
       i += (long long)gridDim.x * 256) {
    float v = w[i];
    s += v * v;
  }
  red[threadIdx.x] = s;
  __syncthreads();
  for (int st = 128; st > 0; st >>= 1) {
    if ((int)threadIdx.x < st) red[threadIdx.x] += red[threadIdx.x + st];
    __syncthreads();
  }
  if (threadIdx.x == 0) partial[blockIdx.x] = red[0];
}

__global__ void finalize_rms(const float* __restrict__ p, float* __restrict__ wstat,
                             float inv_count) {
  if (threadIdx.x == 0 && blockIdx.x == 0) {
    float s1 = 0.f, s2 = 0.f;
    for (int i = 0; i < NPART; ++i) { s1 += p[i]; s2 += p[i + NPART]; }
    const float m1 = fmaxf(s1 * inv_count, EPSQ);
    const float m2 = fmaxf(s2 * inv_count, EPSQ);
    wstat[0] = rsqrtf(m1);
    wstat[1] = rsqrtf(m2);
    wstat[2] = sqrtf(m1);
    wstat[3] = sqrtf(m2);
  }
}

__global__ __launch_bounds__(256) void quant_w(const float* __restrict__ w,
                                               int8_t* __restrict__ q,
                                               const float* __restrict__ wstat,
                                               int idx, long long n) {
  const float rms = wstat[idx];
  for (long long i = (long long)blockIdx.x * 256 + threadIdx.x; i < n;
       i += (long long)gridDim.x * 256) {
    float v = rintf(w[i] * rms);
    v = fminf(1.f, fmaxf(-1.f, v));
    q[i] = (int8_t)v;
  }
}

__global__ __launch_bounds__(256) void quant_rows(const float* __restrict__ x,
                                                  int8_t* __restrict__ q,
                                                  float* __restrict__ inv_scale,
                                                  int ncols) {
  const int row = blockIdx.x;
  const float* xr = x + (size_t)row * ncols;
  int8_t* qr = q + (size_t)row * ncols;
  __shared__ float red[256];
  float mx = 0.f;
  for (int c = threadIdx.x; c < ncols; c += 256) mx = fmaxf(mx, fabsf(xr[c]));
  red[threadIdx.x] = mx;
  __syncthreads();
  for (int st = 128; st > 0; st >>= 1) {
    if ((int)threadIdx.x < st)
      red[threadIdx.x] = fmaxf(red[threadIdx.x], red[threadIdx.x + st]);
    __syncthreads();
  }
  const float scale = 127.0f / fmaxf(red[0], EPSQ);
  for (int c = threadIdx.x; c < ncols; c += 256) {
    float v = rintf(xr[c] * scale);
    v = fminf(127.f, fmaxf(-128.f, v));
    qr[c] = (int8_t)v;
  }
  if (threadIdx.x == 0) inv_scale[row] = 1.0f / scale;
}

#define LDT 80

template <bool GELU>
__global__ __launch_bounds__(256) void gemm_i8_wmma(
    const int8_t* __restrict__ A, const int8_t* __restrict__ W,
    float* __restrict__ C, const float* __restrict__ inv_rowscale,
    const float* __restrict__ wstat, int widx, int M, int N, int K) {
  __shared__ __align__(16) int8_t sA[2][128 * LDT];
  __shared__ __align__(16) int8_t sB[2][128 * LDT];

  const int tid = threadIdx.x;
  const int m0 = blockIdx.y * 128;
  const int n0 = blockIdx.x * 128;

  const int lrow = tid >> 2;
  const int lcol = (tid & 3) * 16;
  const int8_t* gA = A + (size_t)(m0 + lrow) * K + lcol;
  const int8_t* gW = W + (size_t)(n0 + lrow) * K + lcol;
  const size_t rowskip = (size_t)64 * K;

  const int lane = tid & 31;
  const int wave = tid >> 5;
  const int wm = wave >> 1;
  const int wn = wave & 1;
  const int half = lane >> 4;
  const int mn = lane & 15;

  v8i acc[2][4] = {};
  const int nk = K >> 6;
  int buf = 0;

#if USE_ASYNC_LDS
  {
    __builtin_amdgcn_global_load_async_to_lds_b128(
        AS1V4(gA), AS3V4(&sA[0][lrow * LDT + lcol]), 0, 0);
    __builtin_amdgcn_global_load_async_to_lds_b128(
        AS1V4(gA + rowskip), AS3V4(&sA[0][(lrow + 64) * LDT + lcol]), 0, 0);
    __builtin_amdgcn_global_load_async_to_lds_b128(
        AS1V4(gW), AS3V4(&sB[0][lrow * LDT + lcol]), 0, 0);
    __builtin_amdgcn_global_load_async_to_lds_b128(
        AS1V4(gW + rowskip), AS3V4(&sB[0][(lrow + 64) * LDT + lcol]), 0, 0);
    wait_async0();
  }
  __syncthreads();
#else
  int4 ra0 = *(const int4*)gA;
  int4 ra1 = *(const int4*)(gA + rowskip);
  int4 rb0 = *(const int4*)gW;
  int4 rb1 = *(const int4*)(gW + rowskip);
  *(int4*)&sA[0][lrow * LDT + lcol] = ra0;
  *(int4*)&sA[0][(lrow + 64) * LDT + lcol] = ra1;
  *(int4*)&sB[0][lrow * LDT + lcol] = rb0;
  *(int4*)&sB[0][(lrow + 64) * LDT + lcol] = rb1;
#endif

  for (int kt = 0; kt < nk; ++kt) {
#if !USE_ASYNC_LDS
    __syncthreads();
#endif
    const bool more = (kt + 1) < nk;

#if USE_ASYNC_LDS
    if (more) {
      const int nb = buf ^ 1;
      const int8_t* pA = gA + (size_t)(kt + 1) * 64;
      const int8_t* pW = gW + (size_t)(kt + 1) * 64;
      __builtin_amdgcn_global_load_async_to_lds_b128(
          AS1V4(pA), AS3V4(&sA[nb][lrow * LDT + lcol]), 0, 0);
      __builtin_amdgcn_global_load_async_to_lds_b128(
          AS1V4(pA + rowskip), AS3V4(&sA[nb][(lrow + 64) * LDT + lcol]), 0, 0);
      __builtin_amdgcn_global_load_async_to_lds_b128(
          AS1V4(pW), AS3V4(&sB[nb][lrow * LDT + lcol]), 0, 0);
      __builtin_amdgcn_global_load_async_to_lds_b128(
          AS1V4(pW + rowskip), AS3V4(&sB[nb][(lrow + 64) * LDT + lcol]), 0, 0);
    }
#else
    if (more) {
      const int8_t* pA = gA + (size_t)(kt + 1) * 64;
      const int8_t* pW = gW + (size_t)(kt + 1) * 64;
      ra0 = *(const int4*)pA;
      ra1 = *(const int4*)(pA + rowskip);
      rb0 = *(const int4*)pW;
      rb1 = *(const int4*)(pW + rowskip);
    }
#endif

    const int8_t* aT = sA[buf];
    const int8_t* bT = sB[buf];

    v8i af[2];
#pragma unroll
    for (int i = 0; i < 2; ++i) {
      const int8_t* base = aT + (wm * 32 + i * 16 + mn) * LDT + half * 16;
#pragma unroll
      for (int p = 0; p < 2; ++p) {
        int4 t = *(const int4*)(base + p * 32);
        af[i][4 * p] = t.x;
        af[i][4 * p + 1] = t.y;
        af[i][4 * p + 2] = t.z;
        af[i][4 * p + 3] = t.w;
      }
    }
    v8i bf[4];
#pragma unroll
    for (int j = 0; j < 4; ++j) {
      const int8_t* base = bT + (wn * 64 + j * 16 + mn) * LDT + half * 16;
      int4 t0 = *(const int4*)(base);
      int4 t1 = *(const int4*)(base + 32);
      bf[j] = (v8i){t0.x, t0.y, t0.z, t0.w, t1.x, t1.y, t1.z, t1.w};
    }

#pragma unroll
    for (int i = 0; i < 2; ++i)
#pragma unroll
      for (int j = 0; j < 4; ++j)
        acc[i][j] = __builtin_amdgcn_wmma_i32_16x16x64_iu8(
             true, af[i],  true, bf[j], acc[i][j],
             false,  false);

#if USE_ASYNC_LDS
    if (more) wait_async0();
    __syncthreads();
#else
    if (more) {
      const int nb = buf ^ 1;
      *(int4*)&sA[nb][lrow * LDT + lcol] = ra0;
      *(int4*)&sA[nb][(lrow + 64) * LDT + lcol] = ra1;
      *(int4*)&sB[nb][lrow * LDT + lcol] = rb0;
      *(int4*)&sB[nb][(lrow + 64) * LDT + lcol] = rb1;
    }
#endif
    buf ^= 1;
  }

  const float invw = wstat[2 + widx];
#pragma unroll
  for (int i = 0; i < 2; ++i) {
#pragma unroll
    for (int r = 0; r < 8; ++r) {
      const int m = m0 + wm * 32 + i * 16 + half * 8 + r;
      const float s = inv_rowscale[m] * invw;
#pragma unroll
      for (int j = 0; j < 4; ++j) {
        const int n = n0 + wn * 64 + j * 16 + mn;
        float v = (float)acc[i][j][r] * s;
        if (GELU) v = 0.5f * v * (1.0f + erff(v * 0.70710678118654752f));
        C[(size_t)m * N + n] = v;
      }
    }
  }
}

extern "C" void kernel_launch(void* const* d_in, const int* in_sizes, int n_in,
                              void* d_out, int out_size, void* d_ws, size_t ws_size,
                              hipStream_t stream) {
  const int Mrows = 4 * 2048;
  const int D = 2048;
  const int F = 8192;

  const float* x  = (const float*)d_in[0];
  const float* w1 = (const float*)d_in[1];
  const float* w2 = (const float*)d_in[2];
  float* out = (float*)d_out;

  char* ws = (char*)d_ws;
  size_t off = 0;
  auto take = [&](size_t bytes) {
    void* p = ws + off;
    off = (off + bytes + 255) & ~(size_t)255;
    return p;
  };
  float*  wstat    = (float*)take(4 * sizeof(float));
  float*  partials = (float*)take(2 * NPART * sizeof(float));
  float*  inv_sx   = (float*)take((size_t)Mrows * sizeof(float));
  float*  inv_sh   = (float*)take((size_t)Mrows * sizeof(float));
  int8_t* xq       = (int8_t*)take((size_t)Mrows * D);
  int8_t* w1q      = (int8_t*)take((size_t)F * D);
  int8_t* w2q      = (int8_t*)take((size_t)D * F);
  int8_t* hq       = (int8_t*)take((size_t)Mrows * F);
  float*  h        = (float*)take((size_t)Mrows * F * sizeof(float));
  (void)ws_size; (void)n_in; (void)in_sizes; (void)out_size;

  const long long nW = (long long)F * D;

  sumsq_partial<<<NPART, 256, 0, stream>>>(w1, nW, partials);
  sumsq_partial<<<NPART, 256, 0, stream>>>(w2, nW, partials + NPART);
  finalize_rms<<<1, 64, 0, stream>>>(partials, wstat, 1.0f / (float)nW);

  quant_w<<<4096, 256, 0, stream>>>(w1, w1q, wstat, 0, nW);
  quant_w<<<4096, 256, 0, stream>>>(w2, w2q, wstat, 1, nW);
  quant_rows<<<Mrows, 256, 0, stream>>>(x, xq, inv_sx, D);

  dim3 g1(F / 128, Mrows / 128);
  gemm_i8_wmma<true><<<g1, 256, 0, stream>>>(xq, w1q, h, inv_sx, wstat, 0,
                                             Mrows, F, D);

  quant_rows<<<Mrows, 256, 0, stream>>>(h, hq, inv_sh, F);

  dim3 g2(D / 128, Mrows / 128);
  gemm_i8_wmma<false><<<g2, 256, 0, stream>>>(hq, w2q, out, inv_sh, wstat, 1,
                                              Mrows, D, F);
}
